// GraphEncoder_75711683493987
// MI455X (gfx1250) — matrix operand order repaired and checked by reading, not compiled, not run on hardware
//
#include <hip/hip_runtime.h>

// ---------------------------------------------------------------------------
// GraphEncoder forward for MI455X (gfx1250, wave32, WMMA).
// All heavy GEMMs + attention use v_wmma_f32_16x16x32_f16 (f16 in, f32 accum).
// ---------------------------------------------------------------------------

typedef __attribute__((ext_vector_type(16))) _Float16 v16h;
typedef __attribute__((ext_vector_type(8)))  float    v8f;

union FragH {
    v16h v;
    unsigned int u[8];
    _Float16 h[16];
};

#define B_   16
#define N_   64          // N0=63 + virtual node
#define H_   256
#define NH_  8
#define KD_  64
#define FF_  1024
#define L_   4
#define M_   (B_ * N_ * N_)   // 65536 pair-tokens
#define QKN_ (NH_ * KD_)      // 512

__device__ __forceinline__ v8f zero_v8f() {
    v8f z;
#pragma unroll
    for (int r = 0; r < 8; ++r) z[r] = 0.0f;
    return z;
}

// A-fragment K offsets per VGPR pair p (ISA 7.12.2, 16-bit A 16x32)
__device__ __forceinline__ int kkA(int p, int half) {
    return ((p < 4) ? 2 * p : 16 + 2 * (p - 4)) + 8 * half;
}
// B-fragment K offsets per VGPR pair p (16-bit B 32x16; same K order as A)
__device__ __forceinline__ int kkB(int p, int half) {
    return ((p < 4) ? 2 * p : 16 + 2 * (p - 4)) + 8 * half;
}

// ---------------------------------------------------------------------------
// small helpers
// ---------------------------------------------------------------------------
__global__ void f32_to_f16_kernel(const float* __restrict__ in,
                                  _Float16* __restrict__ out, int n) {
    int i = blockIdx.x * blockDim.x + threadIdx.x;
    if (i < n) out[i] = (_Float16)in[i];
}

__global__ void build_mpad_kernel(const float* __restrict__ mask,
                                  float* __restrict__ mpad) {
    int t = blockIdx.x * blockDim.x + threadIdx.x;   // b*64 + i
    if (t >= B_ * N_) return;
    int b = t >> 6, i = t & 63;
    mpad[t] = (i == 0) ? 1.0f : mask[b * 63 + (i - 1)];
}

__device__ __forceinline__ float block_reduce_sum_256(float val, float* red) {
    int c = threadIdx.x;
    red[c] = val;
    __syncthreads();
#pragma unroll
    for (int s = 128; s > 0; s >>= 1) {
        if (c < s) red[c] += red[c + s];
        __syncthreads();
    }
    float r = red[0];
    __syncthreads();
    return r;
}

// ---------------------------------------------------------------------------
// fused fc_in + LayerNorm producing f16 x   (one block per pair-token)
// ---------------------------------------------------------------------------
__global__ __launch_bounds__(256)
void build_x_kernel(const float* __restrict__ nf, const float* __restrict__ ef,
                    const float* __restrict__ W, const float* __restrict__ bias,
                    const float* __restrict__ g, const float* __restrict__ bt,
                    _Float16* __restrict__ X) {
    __shared__ float red[256];
    int tok = blockIdx.x;                 // b*4096 + i*64 + j
    int b = tok >> 12, i = (tok >> 6) & 63, j = tok & 63;
    float e  = (i > 0 && j > 0) ? ef[((size_t)b * 63 + (i - 1)) * 63 + (j - 1)] : 0.0f;
    float ni = (i > 0) ? nf[b * 63 + (i - 1)] : 0.0f;
    float nj = (j > 0) ? nf[b * 63 + (j - 1)] : 0.0f;
    int c = threadIdx.x;
    float v = e * W[c] + ni * W[H_ + c] + nj * W[2 * H_ + c] + bias[c];
    float mean = block_reduce_sum_256(v, red) * (1.0f / H_);
    float d = v - mean;
    float var = block_reduce_sum_256(d * d, red) * (1.0f / H_);
    X[(size_t)tok * H_ + c] = (_Float16)(d * rsqrtf(var + 1e-5f) * g[c] + bt[c]);
}

// ---------------------------------------------------------------------------
// LayerNorm over f32 rows -> f16 x   (one block per row)
// ---------------------------------------------------------------------------
__global__ __launch_bounds__(256)
void ln_kernel(const float* __restrict__ Y, const float* __restrict__ g,
               const float* __restrict__ bt, _Float16* __restrict__ X) {
    __shared__ float red[256];
    size_t row = blockIdx.x;
    int c = threadIdx.x;
    float v = Y[row * H_ + c];
    float mean = block_reduce_sum_256(v, red) * (1.0f / H_);
    float d = v - mean;
    float var = block_reduce_sum_256(d * d, red) * (1.0f / H_);
    X[row * H_ + c] = (_Float16)(d * rsqrtf(var + 1e-5f) * g[c] + bt[c]);
}

// ---------------------------------------------------------------------------
// WMMA GEMM:  C[M,N] = A[M,K] @ Bw[K,N]  (f16 in, f32 accum)
// EPI 0: outH = acc + bias            (f16)
// EPI 1: outH = relu(acc + bias)      (f16)
// EPI 2: outF = acc + bias + R (f16 residual)  (f32, pre-LN)
// block = 256 threads (8 waves); tile 128x64; K step 32, LDS-staged.
// ---------------------------------------------------------------------------
template <int EPI>
__global__ __launch_bounds__(256)
void gemm_kernel(const _Float16* __restrict__ A, const _Float16* __restrict__ Bw,
                 const float* __restrict__ bias, const _Float16* __restrict__ R,
                 _Float16* __restrict__ outH, float* __restrict__ outF,
                 int M, int N, int K) {
    __shared__ _Float16 As[128 * 32];
    __shared__ _Float16 Bs[32 * 64];
    const int tid  = threadIdx.x;
    const int wave = tid >> 5;
    const int lane = tid & 31;
    const int half = lane >> 4;
    const int ln16 = lane & 15;
    const int tileM = blockIdx.x * 128;
    const int tileN = blockIdx.y * 64;

    v8f acc[4];
#pragma unroll
    for (int t = 0; t < 4; ++t) acc[t] = zero_v8f();

    for (int kt = 0; kt < K; kt += 32) {
        // stage A 128x32 (512 chunks of 8 halves)
#pragma unroll
        for (int it = 0; it < 2; ++it) {
            int c = tid + it * 256;
            int row = c >> 2, seg = c & 3;
            *(uint4*)&As[row * 32 + seg * 8] =
                *(const uint4*)&A[(size_t)(tileM + row) * K + kt + seg * 8];
        }
        // stage B 32x64 (256 chunks of 8 halves)
        {
            int row = tid >> 3, seg = tid & 7;
            *(uint4*)&Bs[row * 64 + seg * 8] =
                *(const uint4*)&Bw[(size_t)(kt + row) * N + tileN + seg * 8];
        }
        __syncthreads();

        FragH fa;
        const int ar = wave * 16 + ln16;
#pragma unroll
        for (int p = 0; p < 8; ++p)
            fa.u[p] = *(const unsigned int*)&As[ar * 32 + kkA(p, half)];

#pragma unroll
        for (int ct = 0; ct < 4; ++ct) {
            FragH fb;
            const int col = ct * 16 + ln16;
#pragma unroll
            for (int p = 0; p < 8; ++p) {
                int kk = kkB(p, half);
                fb.h[2 * p]     = Bs[kk * 64 + col];
                fb.h[2 * p + 1] = Bs[(kk + 1) * 64 + col];
            }
            acc[ct] = __builtin_amdgcn_wmma_f32_16x16x32_f16(
                false, fa.v, false, fb.v, (short)0, acc[ct], false, false);
        }
        __syncthreads();
    }

    // epilogue (C/D layout: VGPR r -> row r + 8*half)
    const int rbase = tileM + wave * 16 + 8 * half;
#pragma unroll
    for (int ct = 0; ct < 4; ++ct) {
        int col = tileN + ct * 16 + ln16;
        float bb = bias[col];
#pragma unroll
        for (int r = 0; r < 8; ++r) {
            size_t gidx = (size_t)(rbase + r) * N + col;
            float vv = acc[ct][r] + bb;
            if (EPI == 0) outH[gidx] = (_Float16)vv;
            else if (EPI == 1) outH[gidx] = (_Float16)(vv > 0.0f ? vv : 0.0f);
            else outF[gidx] = vv + (float)R[gidx];
        }
    }
}

// ---------------------------------------------------------------------------
// attention: one block per (b*N+i, head); 128 threads = 4 waves.
// S = Q K^T (WMMA) -> scale + mask bias -> softmax (LDS) -> O = P V (WMMA)
// ---------------------------------------------------------------------------
__global__ __launch_bounds__(128)
void attn_kernel(const _Float16* __restrict__ Q, const _Float16* __restrict__ Km,
                 const _Float16* __restrict__ V, const float* __restrict__ mpad,
                 _Float16* __restrict__ O) {
    __shared__ float    Sf[64 * 64];
    __shared__ _Float16 P[64 * 64];
    __shared__ _Float16 Vs[64 * 64];

    const int bi   = blockIdx.x;          // b*64 + i
    const int b    = bi >> 6;
    const int head = blockIdx.y;
    const size_t base = (size_t)bi * 64;  // first token row of this (b,i)
    const int hoff = head * KD_;

    const int tid  = threadIdx.x;
    const int wave = tid >> 5;
    const int lane = tid & 31;
    const int half = lane >> 4;
    const int ln16 = lane & 15;
    const int j0   = wave * 16;

    // stage V[64 x 64] into LDS
#pragma unroll
    for (int it = 0; it < 4; ++it) {
        int c = tid + it * 128;
        int kk = c >> 3, seg = c & 7;
        *(uint4*)&Vs[kk * 64 + seg * 8] =
            *(const uint4*)&V[(base + kk) * QKN_ + hoff + seg * 8];
    }

    // S = Q K^T  (B fragment lane n holds row n of K -> contiguous pairs)
    v8f acc[4];
#pragma unroll
    for (int t = 0; t < 4; ++t) acc[t] = zero_v8f();
#pragma unroll
    for (int kt = 0; kt < 64; kt += 32) {
        FragH fa;
#pragma unroll
        for (int p = 0; p < 8; ++p)
            fa.u[p] = *(const unsigned int*)
                &Q[(base + j0 + ln16) * QKN_ + hoff + kt + kkA(p, half)];
#pragma unroll
        for (int ct = 0; ct < 4; ++ct) {
            FragH fb;
#pragma unroll
            for (int p = 0; p < 8; ++p)
                fb.u[p] = *(const unsigned int*)
                    &Km[(base + ct * 16 + ln16) * QKN_ + hoff + kt + kkB(p, half)];
            acc[ct] = __builtin_amdgcn_wmma_f32_16x16x32_f16(
                false, fa.v, false, fb.v, (short)0, acc[ct], false, false);
        }
    }

    // scale + key mask bias, write S to LDS
    const float mi = mpad[bi];
#pragma unroll
    for (int ct = 0; ct < 4; ++ct) {
        int kcol = ct * 16 + ln16;
        float kb = (mi * mpad[b * 64 + kcol] > 0.0f) ? 0.0f : -1e9f;
#pragma unroll
        for (int r = 0; r < 8; ++r)
            Sf[(j0 + r + 8 * half) * 64 + kcol] = acc[ct][r] * 0.125f + kb;
    }
    __syncthreads();

    // softmax over k (threads 0..63, one row each)
    if (tid < 64) {
        float mx = -3.4e38f;
        for (int kc = 0; kc < 64; ++kc) mx = fmaxf(mx, Sf[tid * 64 + kc]);
        float s = 0.0f;
        for (int kc = 0; kc < 64; ++kc) {
            float e = __expf(Sf[tid * 64 + kc] - mx);
            Sf[tid * 64 + kc] = e;
            s += e;
        }
        float inv = 1.0f / s;
        for (int kc = 0; kc < 64; ++kc)
            P[tid * 64 + kc] = (_Float16)(Sf[tid * 64 + kc] * inv);
    }
    __syncthreads();

    // O = P x V
    v8f oac[4];
#pragma unroll
    for (int t = 0; t < 4; ++t) oac[t] = zero_v8f();
#pragma unroll
    for (int kt = 0; kt < 64; kt += 32) {
        FragH fa;
#pragma unroll
        for (int p = 0; p < 8; ++p)
            fa.u[p] = *(const unsigned int*)&P[(j0 + ln16) * 64 + kt + kkA(p, half)];
#pragma unroll
        for (int ct = 0; ct < 4; ++ct) {
            FragH fb;
            const int col = ct * 16 + ln16;
#pragma unroll
            for (int p = 0; p < 8; ++p) {
                int kk = kt + kkB(p, half);
                fb.h[2 * p]     = Vs[kk * 64 + col];
                fb.h[2 * p + 1] = Vs[(kk + 1) * 64 + col];
            }
            oac[ct] = __builtin_amdgcn_wmma_f32_16x16x32_f16(
                false, fa.v, false, fb.v, (short)0, oac[ct], false, false);
        }
    }
#pragma unroll
    for (int ct = 0; ct < 4; ++ct) {
#pragma unroll
        for (int r = 0; r < 8; ++r) {
            int row = j0 + r + 8 * half;
            O[(base + row) * QKN_ + hoff + ct * 16 + ln16] = (_Float16)oac[ct][r];
        }
    }
}

// ---------------------------------------------------------------------------
// diagonal extraction -> tuple-flat output: [B,H] (node 0) then [B,63,H]
// ---------------------------------------------------------------------------
__global__ __launch_bounds__(256)
void diag_kernel(const _Float16* __restrict__ X, float* __restrict__ out) {
    int bi = blockIdx.x;                  // b*64 + i
    int b = bi >> 6, i = bi & 63;
    int c = threadIdx.x;
    float v = (float)X[((size_t)(b * N_ + i) * N_ + i) * H_ + c];
    if (i == 0) out[b * H_ + c] = v;
    else out[B_ * H_ + ((size_t)(b * 63 + (i - 1))) * H_ + c] = v;
}

// ---------------------------------------------------------------------------
// launcher
// ---------------------------------------------------------------------------
extern "C" void kernel_launch(void* const* d_in, const int* in_sizes, int n_in,
                              void* d_out, int out_size, void* d_ws, size_t ws_size,
                              hipStream_t stream) {
    const float* nf      = (const float*)d_in[0];
    const float* ef      = (const float*)d_in[1];
    const float* mask    = (const float*)d_in[2];
    const float* fc_in_w = (const float*)d_in[3];
    const float* fc_in_b = (const float*)d_in[4];
    const float* ln_g    = (const float*)d_in[5];
    const float* ln_b    = (const float*)d_in[6];
    const float* Wq      = (const float*)d_in[7];
    const float* bq      = (const float*)d_in[8];
    const float* Wk      = (const float*)d_in[9];
    const float* bk      = (const float*)d_in[10];
    const float* Wv      = (const float*)d_in[11];
    const float* bv      = (const float*)d_in[12];
    const float* Wo      = (const float*)d_in[13];
    const float* bo      = (const float*)d_in[14];
    const float* ln1_g   = (const float*)d_in[15];
    const float* ln1_b   = (const float*)d_in[16];
    const float* W1      = (const float*)d_in[17];
    const float* b1      = (const float*)d_in[18];
    const float* W2      = (const float*)d_in[19];
    const float* b2      = (const float*)d_in[20];
    const float* ln2_g   = (const float*)d_in[21];
    const float* ln2_b   = (const float*)d_in[22];

    char* ws = (char*)d_ws;
    const size_t MB = 1ull << 20;
    _Float16* xf16 = (_Float16*)(ws + 0);         // 32 MB
    float*    yf32 = (float*)(ws + 32 * MB);      // 64 MB
    _Float16* qb   = (_Float16*)(ws + 96 * MB);   // 64 MB
    _Float16* kb   = (_Float16*)(ws + 160 * MB);  // 64 MB
    _Float16* vb   = (_Float16*)(ws + 224 * MB);  // 64 MB
    _Float16* ob   = (_Float16*)(ws + 288 * MB);  // 64 MB
    _Float16* hb   = (_Float16*)(ws + 96 * MB);   // 128 MB, aliases q/k (dead by FFN)
    char* wp = ws + 352 * MB;
    _Float16* wq16 = (_Float16*)wp; wp += (size_t)L_ * H_ * QKN_ * 2;
    _Float16* wk16 = (_Float16*)wp; wp += (size_t)L_ * H_ * QKN_ * 2;
    _Float16* wv16 = (_Float16*)wp; wp += (size_t)L_ * H_ * QKN_ * 2;
    _Float16* wo16 = (_Float16*)wp; wp += (size_t)L_ * QKN_ * H_ * 2;
    _Float16* w116 = (_Float16*)wp; wp += (size_t)L_ * H_ * FF_ * 2;
    _Float16* w216 = (_Float16*)wp; wp += (size_t)L_ * FF_ * H_ * 2;
    float* mpad = (float*)wp;

    const int nQKV = L_ * H_ * QKN_;   // 524288
    const int nWO  = L_ * QKN_ * H_;   // 524288
    const int nW1  = L_ * H_ * FF_;    // 1048576
    const int nW2  = L_ * FF_ * H_;    // 1048576
    f32_to_f16_kernel<<<(nQKV + 255) / 256, 256, 0, stream>>>(Wq, wq16, nQKV);
    f32_to_f16_kernel<<<(nQKV + 255) / 256, 256, 0, stream>>>(Wk, wk16, nQKV);
    f32_to_f16_kernel<<<(nQKV + 255) / 256, 256, 0, stream>>>(Wv, wv16, nQKV);
    f32_to_f16_kernel<<<(nWO  + 255) / 256, 256, 0, stream>>>(Wo, wo16, nWO);
    f32_to_f16_kernel<<<(nW1  + 255) / 256, 256, 0, stream>>>(W1, w116, nW1);
    f32_to_f16_kernel<<<(nW2  + 255) / 256, 256, 0, stream>>>(W2, w216, nW2);
    build_mpad_kernel<<<4, 256, 0, stream>>>(mask, mpad);

    build_x_kernel<<<M_, 256, 0, stream>>>(nf, ef, fc_in_w, fc_in_b, ln_g, ln_b, xf16);

    for (int l = 0; l < L_; ++l) {
        gemm_kernel<0><<<dim3(M_ / 128, QKN_ / 64), 256, 0, stream>>>(
            xf16, wq16 + (size_t)l * H_ * QKN_, bq + l * QKN_, nullptr,
            qb, nullptr, M_, QKN_, H_);
        gemm_kernel<0><<<dim3(M_ / 128, QKN_ / 64), 256, 0, stream>>>(
            xf16, wk16 + (size_t)l * H_ * QKN_, bk + l * QKN_, nullptr,
            kb, nullptr, M_, QKN_, H_);
        gemm_kernel<0><<<dim3(M_ / 128, QKN_ / 64), 256, 0, stream>>>(
            xf16, wv16 + (size_t)l * H_ * QKN_, bv + l * QKN_, nullptr,
            vb, nullptr, M_, QKN_, H_);

        attn_kernel<<<dim3(B_ * N_, NH_), 128, 0, stream>>>(qb, kb, vb, mpad, ob);

        gemm_kernel<2><<<dim3(M_ / 128, H_ / 64), 256, 0, stream>>>(
            ob, wo16 + (size_t)l * QKN_ * H_, bo + l * H_, xf16,
            nullptr, yf32, M_, H_, QKN_);
        ln_kernel<<<M_, 256, 0, stream>>>(yf32, ln1_g + l * H_, ln1_b + l * H_, xf16);

        gemm_kernel<1><<<dim3(M_ / 128, FF_ / 64), 256, 0, stream>>>(
            xf16, w116 + (size_t)l * H_ * FF_, b1 + l * FF_, nullptr,
            hb, nullptr, M_, FF_, H_);
        gemm_kernel<2><<<dim3(M_ / 128, H_ / 64), 256, 0, stream>>>(
            hb, w216 + (size_t)l * FF_ * H_, b2 + l * H_, xf16,
            nullptr, yf32, M_, H_, FF_);
        ln_kernel<<<M_, 256, 0, stream>>>(yf32, ln2_g + l * H_, ln2_b + l * H_, xf16);
    }

    diag_kernel<<<B_ * N_, 256, 0, stream>>>(xf16, (float*)d_out);
}
